// VITS_84439057040006
// MI455X (gfx1250) — compile-verified, matrix operand order checked by reading, not run on hardware
//
#include <hip/hip_runtime.h>
#include <hip/hip_bf16.h>
#include <stdint.h>

typedef _Float16 v16h __attribute__((ext_vector_type(16)));
typedef _Float16 v8h  __attribute__((ext_vector_type(8)));
typedef float    v8f  __attribute__((ext_vector_type(8)));

#define BB 8
#define DD 192
#define TXX 1024
#define TYY 4096
#define KP  (2*DD)          // 384: k<D -> (z, mean*s) ; k>=D -> (-0.5 z^2, s)
#define KSTEPS (KP/32)      // 12
#define NEGF (-1.0e9f)
#define HALF_LOG_2PI 0.9189385332046727f

// ---------------- prep B-side: f16 [B][Tx][KP] K-contiguous + ncx[b][x] ----------------
__global__ __launch_bounds__(256) void prep_b_kernel(const float* __restrict__ xm,
                                                     const float* __restrict__ xs,
                                                     _Float16* __restrict__ Bf,
                                                     float* __restrict__ ncx) {
  const int x = blockIdx.x, b = blockIdx.y;
  __shared__ float red[256];
  float acc = 0.f;
  for (int d = threadIdx.x; d < DD; d += 256) {
    size_t o = ((size_t)(b*DD + d))*TXX + x;
    float m = xm[o], lg = xs[o];
    float s = __expf(-2.f * lg);
    size_t ob = ((size_t)(b*TXX + x))*KP;
    Bf[ob + d]      = (_Float16)(m * s);   // pairs with z
    Bf[ob + DD + d] = (_Float16)s;         // pairs with -0.5 z^2
    acc += -lg - 0.5f * m * m * s;         // nc1 + nc4 contribution
  }
  red[threadIdx.x] = acc; __syncthreads();
  for (int s = 128; s > 0; s >>= 1) {
    if (threadIdx.x < s) red[threadIdx.x] += red[threadIdx.x + s];
    __syncthreads();
  }
  if (threadIdx.x == 0) ncx[b*TXX + x] = red[0] - HALF_LOG_2PI * (float)DD;
}

// ---------------- prep A-side: f16 [B][Ty][KP] K-contiguous ----------------
__global__ void prep_a_kernel(const float* __restrict__ z, _Float16* __restrict__ Af) {
  size_t i = (size_t)blockIdx.x * 256 + threadIdx.x;
  if (i >= (size_t)BB * TYY * DD) return;
  int d = (int)(i % DD);
  size_t r = i / DD;
  int y = (int)(r % TYY);
  int b = (int)(r / TYY);
  float zv = z[((size_t)(b*DD + d))*TYY + y];
  size_t oa = ((size_t)(b*TYY + y))*KP;
  Af[oa + d]      = (_Float16)zv;
  Af[oa + DD + d] = (_Float16)(-0.5f * zv * zv);
}

// ---------------- WMMA GEMM: neg_cent[b][y][x] = A(y,:)·B(:,x) + ncx[b][x] ----------------
// One wave -> 16(M) x 64(N) strip, 4 accumulators, K-loop 384 in steps of 32.
// Double-buffered fragment loads so WMMAs overlap the next K-step's global loads.
__global__ __launch_bounds__(256) void gemm_negcent_kernel(const _Float16* __restrict__ A,
                                                           const _Float16* __restrict__ Bm,
                                                           const float* __restrict__ ncx,
                                                           float* __restrict__ neg_cent) {
  const int lane = threadIdx.x & 31;
  const int wave = threadIdx.x >> 5;
  const int b  = blockIdx.z;
  const int x0 = blockIdx.x * 64;
  const int y0 = (blockIdx.y * 8 + wave) * 16;

  const int row  = lane & 15;        // A row / B col / D col for this lane
  const int base = (lane >> 4) * 8;  // half-wave K offset per ISA fragment layout

  const _Float16* ap = A + ((size_t)(b*TYY + y0 + row))*KP + base;
  const _Float16* bp[4];
  #pragma unroll
  for (int n = 0; n < 4; ++n)
    bp[n] = Bm + ((size_t)(b*TXX + x0 + n*16 + row))*KP + base;

  v8f acc[4] = {};
  v8h alo[2], ahi[2], blo[4][2], bhi[4][2];

  // prologue: K-step 0 into buffer 0
  alo[0] = *(const v8h*)(ap);
  ahi[0] = *(const v8h*)(ap + 16);
  #pragma unroll
  for (int n = 0; n < 4; ++n) {
    blo[n][0] = *(const v8h*)(bp[n]);
    bhi[n][0] = *(const v8h*)(bp[n] + 16);
  }

  #pragma unroll
  for (int ks = 0; ks < KSTEPS; ++ks) {
    const int cur = ks & 1, nxt = cur ^ 1;
    if (ks + 1 < KSTEPS) {                     // issue next K-step's loads first
      const int k1 = (ks + 1) * 32;
      alo[nxt] = *(const v8h*)(ap + k1);
      ahi[nxt] = *(const v8h*)(ap + k1 + 16);
      #pragma unroll
      for (int n = 0; n < 4; ++n) {
        blo[n][nxt] = *(const v8h*)(bp[n] + k1);
        bhi[n][nxt] = *(const v8h*)(bp[n] + k1 + 16);
      }
    }
    v16h a = __builtin_shufflevector(alo[cur], ahi[cur],
                                     0,1,2,3,4,5,6,7,8,9,10,11,12,13,14,15);
    #pragma unroll
    for (int n = 0; n < 4; ++n) {
      v16h f = __builtin_shufflevector(blo[n][cur], bhi[n][cur],
                                       0,1,2,3,4,5,6,7,8,9,10,11,12,13,14,15);
      acc[n] = __builtin_amdgcn_wmma_f32_16x16x32_f16(false, a, false, f,
                                                      (short)0, acc[n], false, false);
    }
  }

  // Epilogue: D layout -> lane holds N=lane&15, rows mbase..mbase+7 in acc[][r]
  const int mbase = (lane >> 4) * 8;
  #pragma unroll
  for (int n = 0; n < 4; ++n) {
    const int xcol = x0 + n*16 + (lane & 15);
    const float add = ncx[b*TXX + xcol];
    #pragma unroll
    for (int r = 0; r < 8; ++r) {
      neg_cent[((size_t)b*TYY + (y0 + mbase + r))*TXX + xcol] = acc[n][r] + add;
    }
  }
}

// ---------------- MAS forward: 1 block per batch, Tx lanes, dirs packed via ballot ----------------
__global__ __launch_bounds__(1024) void mas_forward_kernel(const float* __restrict__ nc,
                                                           const int* __restrict__ xl,
                                                           uint32_t* __restrict__ dirs) {
  const int b = blockIdx.x;
  const int x = threadIdx.x;
  const int lane = x & 31, wave = x >> 5;
  const int xlen = xl[b];
  __shared__ float sb[32];

  float v = (x < xlen ? nc[(size_t)b*TYY*TXX + x] : NEGF) + (x == 0 ? 0.f : NEGF);

  for (int t = 1; t < TYY; ++t) {
    // hide the next row's fetch behind this step's barriers/compares
    if (t + 1 < TYY)
      __builtin_prefetch((const void*)(nc + ((size_t)(b*TYY + t + 1))*TXX + x), 0, 1);
    if (lane == 31) sb[wave] = v;
    __syncthreads();
    float left = __shfl_up(v, 1, 32);
    if (lane == 0) left = (wave == 0) ? NEGF : sb[wave - 1];
    const bool dir = left > v;                       // from_diag = shifted > v
    unsigned long long m = __ballot(dir ? 1 : 0);    // wave32 -> low 32 bits
    if (lane == 0) dirs[((size_t)(b*TYY + t))*32 + wave] = (uint32_t)m;
    float c = nc[((size_t)(b*TYY + t))*TXX + x];
    if (x >= xlen) c = NEGF;
    v = c + fmaxf(v, left);
    __syncthreads();
  }
}

// ---------------- backtrack: path[b][t] = chosen x (or -1 when t >= y_len) ----------------
__global__ void backtrack_kernel(const uint32_t* __restrict__ dirs,
                                 const int* __restrict__ xl, const int* __restrict__ yl,
                                 int* __restrict__ path) {
  int b = threadIdx.x;
  if (b >= BB) return;
  const int ylen = yl[b];
  int idx = xl[b] - 1;
  for (int t = TYY - 1; t >= 0; --t) {
    if (t < ylen) {
      path[b*TYY + t] = idx;
      int dec = 0;
      if (t > 0) {
        uint32_t w = dirs[((size_t)(b*TYY + t))*32 + (idx >> 5)];
        dec = (int)((w >> (idx & 31)) & 1u);
      }
      idx -= dec;
    } else {
      path[b*TYY + t] = -1;
    }
  }
}

// ---------------- outputs ----------------
__global__ void attn_fill_kernel(const int* __restrict__ path, float* __restrict__ attn) {
  size_t i = (size_t)blockIdx.x * 256 + threadIdx.x;
  if (i >= (size_t)BB * TYY * TXX) return;
  int x = (int)(i % TXX);
  size_t r = i / TXX;
  int t = (int)(r % TYY);
  int b = (int)(r / TYY);
  float val = (path[b*TYY + t] == x) ? 1.0f : 0.0f;
  __builtin_nontemporal_store(val, &attn[i]);   // write-once 128MB: keep out of L2
}

__global__ void w_zero_kernel(float* __restrict__ w) {
  int i = blockIdx.x * 256 + threadIdx.x;
  if (i < BB * TXX) w[i] = 0.f;
}

__global__ void w_count_kernel(const int* __restrict__ path, float* __restrict__ w) {
  int i = blockIdx.x * 256 + threadIdx.x;
  if (i >= BB * TYY) return;
  int j = path[i];
  if (j >= 0) atomicAdd(&w[(i / TYY)*TXX + j], 1.0f);
}

__global__ __launch_bounds__(256) void l_length_kernel(const float* __restrict__ logw,
                                                       const float* __restrict__ w,
                                                       const int* __restrict__ xl,
                                                       float* __restrict__ ll) {
  __shared__ float red[256];
  const int b = blockIdx.x;
  float tot = 0.f;
  for (int i = 0; i < BB; ++i) tot += (float)xl[i];   // jnp.sum(x_mask) over whole batch
  float acc = 0.f;
  for (int x = threadIdx.x; x < TXX; x += 256) {
    float d = logw[b*TXX + x] - logf(w[b*TXX + x] + 1e-6f);
    acc += d * d;
  }
  red[threadIdx.x] = acc; __syncthreads();
  for (int s = 128; s > 0; s >>= 1) {
    if (threadIdx.x < s) red[threadIdx.x] += red[threadIdx.x + s];
    __syncthreads();
  }
  if (threadIdx.x == 0) ll[b] = red[0] / tot;
}

__global__ void gather_exp_kernel(const int* __restrict__ path,
                                  const float* __restrict__ xm, const float* __restrict__ xs,
                                  float* __restrict__ om, float* __restrict__ ol) {
  const int y = blockIdx.x * 256 + threadIdx.x;
  const int d = blockIdx.y, b = blockIdx.z;
  const int j = path[b*TYY + y];
  float m = 0.f, l = 0.f;
  if (j >= 0) {
    size_t o = ((size_t)(b*DD + d))*TXX + j;
    m = xm[o]; l = xs[o];
  }
  size_t oo = ((size_t)(b*DD + d))*TYY + y;
  __builtin_nontemporal_store(m, &om[oo]);      // write-once outputs
  __builtin_nontemporal_store(l, &ol[oo]);
}

// ---------------- launch ----------------
extern "C" void kernel_launch(void* const* d_in, const int* in_sizes, int n_in,
                              void* d_out, int out_size, void* d_ws, size_t ws_size,
                              hipStream_t stream) {
  (void)in_sizes; (void)n_in; (void)out_size; (void)ws_size;
  const float* z      = (const float*)d_in[0];   // [B,D,Ty]
  const float* x_mean = (const float*)d_in[1];   // [B,D,Tx]
  const float* x_logs = (const float*)d_in[2];   // [B,D,Tx]
  const float* logw   = (const float*)d_in[3];   // [B,1,Tx]
  const int*   x_len  = (const int*)d_in[4];     // [B]
  const int*   y_len  = (const int*)d_in[5];     // [B]

  // workspace carve-up (all offsets multiples of 256B)
  char* ws = (char*)d_ws;
  size_t off = 0;
  float*     neg_cent = (float*)(ws + off);     off += (size_t)BB*TYY*TXX*sizeof(float);    // 128 MB
  _Float16*  Af       = (_Float16*)(ws + off);  off += (size_t)BB*TYY*KP*sizeof(_Float16);  // 24 MB
  _Float16*  Bf       = (_Float16*)(ws + off);  off += (size_t)BB*TXX*KP*sizeof(_Float16);  // 6 MB
  float*     ncx      = (float*)(ws + off);     off += (size_t)BB*TXX*sizeof(float);
  uint32_t*  dirs     = (uint32_t*)(ws + off);  off += (size_t)BB*TYY*32*sizeof(uint32_t);  // 4 MB
  int*       path     = (int*)(ws + off);       off += (size_t)BB*TYY*sizeof(int);

  // output layout (flat, return order): l_length, attn, x_mean_exp, x_logs_exp, w
  float* out = (float*)d_out;
  const size_t O_LL   = 0;
  const size_t O_ATTN = O_LL   + BB;
  const size_t O_MEAN = O_ATTN + (size_t)BB*TYY*TXX;
  const size_t O_LOGS = O_MEAN + (size_t)BB*DD*TYY;
  const size_t O_W    = O_LOGS + (size_t)BB*DD*TYY;

  prep_b_kernel<<<dim3(TXX, BB), 256, 0, stream>>>(x_mean, x_logs, Bf, ncx);
  prep_a_kernel<<<(int)(((size_t)BB*TYY*DD + 255)/256), 256, 0, stream>>>(z, Af);

  gemm_negcent_kernel<<<dim3(TXX/64, TYY/16/8, BB), 256, 0, stream>>>(Af, Bf, ncx, neg_cent);

  mas_forward_kernel<<<BB, TXX, 0, stream>>>(neg_cent, x_len, dirs);
  backtrack_kernel<<<1, BB, 0, stream>>>(dirs, x_len, y_len, path);

  attn_fill_kernel<<<(int)(((size_t)BB*TYY*TXX)/256), 256, 0, stream>>>(path, out + O_ATTN);
  w_zero_kernel<<<(BB*TXX + 255)/256, 256, 0, stream>>>(out + O_W);
  w_count_kernel<<<(BB*TYY + 255)/256, 256, 0, stream>>>(path, out + O_W);
  l_length_kernel<<<BB, 256, 0, stream>>>(logw, out + O_W, x_len, out + O_LL);
  gather_exp_kernel<<<dim3(TYY/256, DD, BB), 256, 0, stream>>>(path, x_mean, x_logs,
                                                               out + O_MEAN, out + O_LOGS);
}
